// MultiheadAttentionRPE_47373489275155
// MI455X (gfx1250) — matrix operand order repaired and checked by reading, not compiled, not run on hardware
//
#include <hip/hip_runtime.h>
#include <hip/hip_bf16.h>
#include <math.h>

// Problem constants (fixed by the reference)
#define TT 1536   // sequence length
#define HH 8      // heads
#define DK 64     // key dim per head
#define DV 192    // value dim per head
#define EE 1536   // embedding (H*V)
#define CC 1536   // input channels
#define MP 3072   // padded relative-position rows (2T-1 = 3071 -> 3072)

typedef __attribute__((ext_vector_type(16))) __bf16 v16bf;
typedef __attribute__((ext_vector_type(8)))  float  v8f;

struct __attribute__((aligned(16))) U128 { unsigned int u[4]; };
union Frag16 { U128 q[2]; v16bf v; };

// float -> bf16 round-to-nearest-even
__device__ inline __bf16 f2bf(float f) {
  union { float f; unsigned u; } a; a.f = f;
  unsigned r = a.u + 0x7FFFu + ((a.u >> 16) & 1u);
  unsigned short h = (unsigned short)(r >> 16);
  union { unsigned short s; __bf16 b; } o; o.s = h;
  return o.b;
}

__device__ inline v8f wmma_bf16(v16bf a, v16bf b, v8f c) {
  // (neg_a, A, neg_b, B, c_mod, C, reuse_a, reuse_b)
  return __builtin_amdgcn_wmma_f32_16x16x32_bf16(false, a, false, b, (short)0, c,
                                                 false, false);
}

// A-fragment 16x32 (MxK) from row-major [row, k] source.
// lane<16: row=lane, K={k0..k0+7, k0+16..k0+23}; lane>=16: K shifted by 8.
__device__ inline v16bf afrag(const __bf16* base, int stride, int row0, int k0) {
  int lane = threadIdx.x & 31;
  const __bf16* p = base + (size_t)(row0 + (lane & 15)) * stride + k0
                    + ((lane & 16) ? 8 : 0);
  Frag16 f;
  f.q[0] = *(const U128*)p;
  f.q[1] = *(const U128*)(p + 16);
  return f.v;
}

// B-fragment 32x16 (KxN) from row-major [n, k] source (i.e. B^T rows).
// lane<16: col n=lane, K={k0..k0+7, k0+16..k0+23}; lane>=16: K shifted by 8.
__device__ inline v16bf bfrag(const __bf16* base, int stride, int n0, int k0) {
  int lane = threadIdx.x & 31;
  const __bf16* p = base + (size_t)(n0 + (lane & 15)) * stride + k0
                    + ((lane & 16) ? 8 : 0);
  Frag16 f;
  f.q[0] = *(const U128*)p;
  f.q[1] = *(const U128*)(p + 16);
  return f.v;
}

// Async global->LDS 16-byte copy (CDNA5 TDM-adjacent path, ASYNCcnt-tracked).
// INST_OFFSET applies to both the LDS and the global address.
__device__ inline void async_cp16(unsigned lds_off, const __bf16* g) {
  asm volatile("global_load_async_to_lds_b128 %0, %1, off"
               :: "v"(lds_off), "v"(g) : "memory");
}
__device__ inline void async_cp16_o16(unsigned lds_off, const __bf16* g) {
  asm volatile("global_load_async_to_lds_b128 %0, %1, off offset:16"
               :: "v"(lds_off), "v"(g) : "memory");
}

// ---------------------------------------------------------------- casts

__global__ void cast_bf16_kernel(const float* __restrict__ src,
                                 __bf16* __restrict__ dst, int n) {
  int i = blockIdx.x * blockDim.x + threadIdx.x;
  if (i < n) dst[i] = f2bf(src[i]);
}

// ------------------------------------------- positional features (Enformer)

__device__ inline float gamma_logp(float ap, int i) {
  // mean = 48*(i+1), stddev = 24  => conc = 4*(i+1)^2, rate = (i+1)/12
  float mean = 48.0f * (float)(i + 1);
  const float sd = 24.0f;
  float conc = (mean / sd) * (mean / sd);
  float rate = mean / (sd * sd);
  return (conc - 1.0f) * __logf(ap) - rate * ap
         - (lgammaf(conc) - conc * __logf(rate));
}

__global__ void gamma_max_kernel(float* __restrict__ out) {
  __shared__ float red[256];
  float mx = 0.0f;
  for (int idx = threadIdx.x; idx < 3071 * 32; idx += 256) {
    int m = idx >> 5, i = idx & 31;
    float ap = fabsf((float)(m - (TT - 1)));
    float p = __expf(gamma_logp(ap, i)) + 1e-8f;
    mx = fmaxf(mx, p);
  }
  red[threadIdx.x] = mx;
  __syncthreads();
  for (int s = 128; s > 0; s >>= 1) {
    if ((int)threadIdx.x < s)
      red[threadIdx.x] = fmaxf(red[threadIdx.x], red[threadIdx.x + s]);
    __syncthreads();
  }
  if (threadIdx.x == 0) out[0] = red[0];
}

__device__ inline float emb_val(float ap, int c, float invmax) {
  if (c < 32) {                      // exponential decay
    float li = 3.0f + (float)c * ((10.58496250072f - 3.0f) / 31.0f);
    float hl = exp2f(li);
    return exp2f(-ap / hl);
  } else if (c < 64) {               // central mask
    float width = exp2f((float)(c - 32 + 1)) - 1.0f;
    return (width > ap) ? 1.0f : 0.0f;
  } else {                           // gamma pdf, globally normalized
    float prob = __expf(gamma_logp(ap, c - 64)) + 1e-8f;
    return prob * invmax;
  }
}

__global__ void pe_fill_kernel(__bf16* __restrict__ pe,
                               const float* __restrict__ maxp) {
  int gid = blockIdx.x * blockDim.x + threadIdx.x;
  if (gid >= MP * 192) return;
  int m = gid / 192, f = gid % 192;
  float val = 0.0f;
  if (m < 2 * TT - 1) {
    float pos = (float)(m - (TT - 1));
    float ap = fabsf(pos);
    int c = (f < 96) ? f : f - 96;
    float e = emb_val(ap, c, 1.0f / maxp[0]);
    if (f < 96) {
      val = e;
    } else {
      float sgn = (pos > 0.0f) ? 1.0f : ((pos < 0.0f) ? -1.0f : 0.0f);
      val = sgn * e;
    }
  }
  pe[gid] = f2bf(val);
}

// ------------------------------------------------ mode-switched WMMA GEMM
// C[M,N] = A[M,Kd] * B[N,Kd]^T, bf16 inputs, fp32 accumulate.
// block = 256 threads (8 waves), tile 128x128, k-step 32.
// Double-buffered LDS filled by async global->LDS copies (ASYNCcnt),
// one barrier per k-step.

enum { MODE_QCB = 0, MODE_K = 1, MODE_VT = 2, MODE_RK = 3, MODE_FINAL = 4 };

__global__ __launch_bounds__(256) void gemm_bf16_kernel(
    const __bf16* __restrict__ A, const __bf16* __restrict__ Bm,
    int Kd, int mode,
    __bf16* __restrict__ out0, __bf16* __restrict__ out1,
    float* __restrict__ outf,
    const float* __restrict__ bias0, const float* __restrict__ bias1) {
  __shared__ __attribute__((aligned(16))) __bf16 As[2][128][40];
  __shared__ __attribute__((aligned(16))) __bf16 Bs[2][128][40];
  const int tid = threadIdx.x;
  const int lane = tid & 31;
  const int w = tid >> 5;
  const int mw = w >> 1;   // 0..3 : 32 rows each
  const int nw = w & 1;    // 0..1 : 64 cols each
  const int m0 = blockIdx.y * 128;
  const int n0 = blockIdx.x * 128;

  const v8f vz = {};
  v8f acc[2][4];
  for (int i = 0; i < 2; ++i)
    for (int j = 0; j < 4; ++j) acc[i][j] = vz;

  // Each thread owns a 32-byte chunk of the A tile and of the B tile.
  const int srow = tid >> 1;
  const int scol = (tid & 1) * 16;
  const __bf16* ga = A + (size_t)(m0 + srow) * Kd + scol;
  const __bf16* gb = Bm + (size_t)(n0 + srow) * Kd + scol;
  const unsigned lda0 = (unsigned)(size_t)(void*)&As[0][srow][scol];
  const unsigned ldb0 = (unsigned)(size_t)(void*)&Bs[0][srow][scol];
  const unsigned bufstep = (unsigned)sizeof(As[0]);  // 128*40*2 bytes

  const int nsteps = Kd >> 5;

  // prologue: stage 0 -> buffer 0 (4 async copies, ASYNCcnt += 4)
  async_cp16(lda0, ga);
  async_cp16_o16(lda0, ga);
  async_cp16(ldb0, gb);
  async_cp16_o16(ldb0, gb);

  for (int s = 0; s < nsteps; ++s) {
    const int cur = s & 1;
    if (s + 1 < nsteps) {
      const __bf16* gan = ga + (size_t)(s + 1) * 32;
      const __bf16* gbn = gb + (size_t)(s + 1) * 32;
      const unsigned la = lda0 + (unsigned)(cur ^ 1) * bufstep;
      const unsigned lb = ldb0 + (unsigned)(cur ^ 1) * bufstep;
      async_cp16(la, gan);
      async_cp16_o16(la, gan);
      async_cp16(lb, gbn);
      async_cp16_o16(lb, gbn);
      // 4 newly issued outstanding allowed; async loads complete in order,
      // so this guarantees stage s has fully landed in LDS.
      asm volatile("s_wait_asynccnt 0x4" ::: "memory");
    } else {
      asm volatile("s_wait_asynccnt 0x0" ::: "memory");
    }
    __syncthreads();

    v16bf af0 = afrag(&As[cur][0][0], 40, mw * 32, 0);
    v16bf af1 = afrag(&As[cur][0][0], 40, mw * 32 + 16, 0);
    v16bf b0 = bfrag(&Bs[cur][0][0], 40, nw * 64, 0);
    v16bf b1 = bfrag(&Bs[cur][0][0], 40, nw * 64 + 16, 0);
    v16bf b2 = bfrag(&Bs[cur][0][0], 40, nw * 64 + 32, 0);
    v16bf b3 = bfrag(&Bs[cur][0][0], 40, nw * 64 + 48, 0);
    acc[0][0] = wmma_bf16(af0, b0, acc[0][0]);
    acc[0][1] = wmma_bf16(af0, b1, acc[0][1]);
    acc[0][2] = wmma_bf16(af0, b2, acc[0][2]);
    acc[0][3] = wmma_bf16(af0, b3, acc[0][3]);
    acc[1][0] = wmma_bf16(af1, b0, acc[1][0]);
    acc[1][1] = wmma_bf16(af1, b1, acc[1][1]);
    acc[1][2] = wmma_bf16(af1, b2, acc[1][2]);
    acc[1][3] = wmma_bf16(af1, b3, acc[1][3]);
  }

  // Epilogue. C layout: vgpr r -> row r (lanes 0-15) / row r+8 (lanes 16-31),
  // column = lane & 15.
  const int hi8 = (lane & 16) ? 8 : 0;
  const int cl = lane & 15;
  for (int i = 0; i < 2; ++i) {
    for (int j = 0; j < 4; ++j) {
      const int gmb = m0 + mw * 32 + i * 16 + hi8;
      const int gn = n0 + nw * 64 + j * 16 + cl;
      for (int r = 0; r < 8; ++r) {
        const int gm = gmb + r;
        const float v = acc[i][j][r];
        if (mode == MODE_QCB) {
          int b = gm / TT, t = gm - b * TT;
          int h = gn >> 6, kk = gn & 63;
          size_t o = ((((size_t)(b * HH + h)) * TT + t) << 6) | (unsigned)kk;
          float s = v * 0.125f;  // K^-0.5
          out0[o] = f2bf(s + bias0[gn]);   // + r_w_bias
          out1[o] = f2bf(s + bias1[gn]);   // + r_r_bias
        } else if (mode == MODE_K) {
          int b = gm / TT, t = gm - b * TT;
          int h = gn >> 6, kk = gn & 63;
          size_t o = ((((size_t)(b * HH + h)) * TT + t) << 6) | (unsigned)kk;
          out0[o] = f2bf(v);
        } else if (mode == MODE_VT) {       // V stored transposed: [bh][vv][t]
          int b = gm / TT, t = gm - b * TT;
          int h = gn / DV, vv = gn - h * DV;
          size_t o = (((size_t)(b * HH + h)) * DV + vv) * TT + t;
          out0[o] = f2bf(v);
        } else if (mode == MODE_RK) {       // r_k: [h][m][kk], m padded to 3072
          int h = gn >> 6, kk = gn & 63;
          size_t o = (((size_t)h * MP + gm) << 6) | (unsigned)kk;
          out0[o] = f2bf(v);
        } else {                            // FINAL: fp32 out + be
          outf[(size_t)gm * EE + gn] = v + bias0[gn];
        }
      }
    }
  }
}

// --------------------------------------- fused flash attention with RPE
// Grid: (T/128, B*H). Block: 256 threads = 8 waves; each wave owns 16 q-rows.
// shifted_rel[q,j] = u[q, (T-1)+j-q]; per 32-key step we compute the 3 aligned
// 16-wide u tiles covering the needed diagonal band and gather via LDS.

__global__ __launch_bounds__(256) void attn_rpe_kernel(
    const __bf16* __restrict__ qcb, const __bf16* __restrict__ qrb,
    const __bf16* __restrict__ kmat, const __bf16* __restrict__ vt,
    const __bf16* __restrict__ rk, __bf16* __restrict__ ao) {
  __shared__ float uband[8][16][49];
  __shared__ __attribute__((aligned(16))) __bf16 parr[8][16][32];

  const int bh = blockIdx.y;
  const int b = bh >> 3, h = bh & 7;
  const int w = threadIdx.x >> 5;
  const int lane = threadIdx.x & 31;
  const int q0 = blockIdx.x * 128 + w * 16;
  const int hi8 = (lane & 16) ? 8 : 0;
  const int cl = lane & 15;

  const __bf16* qc_p = qcb + (size_t)bh * TT * DK;
  const __bf16* qr_p = qrb + (size_t)bh * TT * DK;
  const __bf16* k_p = kmat + (size_t)bh * TT * DK;
  const __bf16* vt_p = vt + (size_t)bh * DV * TT;
  const __bf16* rk_p = rk + (size_t)h * MP * DK;

  const v16bf qc0 = afrag(qc_p, DK, q0, 0);
  const v16bf qc1 = afrag(qc_p, DK, q0, 32);
  const v16bf qr0 = afrag(qr_p, DK, q0, 0);
  const v16bf qr1 = afrag(qr_p, DK, q0, 32);

  const v8f vz = {};
  v8f o[12];
#pragma unroll
  for (int t = 0; t < 12; ++t) o[t] = vz;
  float mrun[8], lrun[8];
#pragma unroll
  for (int r = 0; r < 8; ++r) { mrun[r] = -1e30f; lrun[r] = 0.0f; }

  for (int j0 = 0; j0 < TT; j0 += 32) {
    // content logits: (q*s + r_w_bias) . k
    v8f s0 = vz, s1 = vz;
    s0 = wmma_bf16(qc0, bfrag(k_p, DK, j0, 0), s0);
    s0 = wmma_bf16(qc1, bfrag(k_p, DK, j0, 32), s0);
    s1 = wmma_bf16(qc0, bfrag(k_p, DK, j0 + 16, 0), s1);
    s1 = wmma_bf16(qc1, bfrag(k_p, DK, j0 + 16, 32), s1);

    // rel band: u tiles at m0a, m0a+16, m0a+32 (m0a in [0, 3024])
    const int m0a = (TT - 16) + j0 - q0;
#pragma unroll
    for (int t = 0; t < 3; ++t) {
      v8f u = vz;
      u = wmma_bf16(qr0, bfrag(rk_p, DK, m0a + t * 16, 0), u);
      u = wmma_bf16(qr1, bfrag(rk_p, DK, m0a + t * 16, 32), u);
#pragma unroll
      for (int r = 0; r < 8; ++r)
        uband[w][r + hi8][t * 16 + cl] = u[r];
    }

    // gather shifted rel, online softmax over the 32 keys
    float scale[8];
#pragma unroll
    for (int r = 0; r < 8; ++r) {
      const int row = r + hi8;
      const int c = cl - row + 15;                 // in [0, 30]
      float a0 = s0[r] + uband[w][row][c];
      float a1 = s1[r] + uband[w][row][c + 16];    // in [16, 46]
      float vmax = fmaxf(a0, a1);
      for (int msk = 1; msk <= 8; msk <<= 1)
        vmax = fmaxf(vmax, __shfl_xor(vmax, msk, 32));
      float mnew = fmaxf(mrun[r], vmax);
      scale[r] = __expf(mrun[r] - mnew);
      float p0 = __expf(a0 - mnew);
      float p1 = __expf(a1 - mnew);
      float rs = p0 + p1;
      for (int msk = 1; msk <= 8; msk <<= 1)
        rs += __shfl_xor(rs, msk, 32);
      lrun[r] = lrun[r] * scale[r] + rs;
      mrun[r] = mnew;
      parr[w][row][cl] = f2bf(p0);
      parr[w][row][16 + cl] = f2bf(p1);
    }

#pragma unroll
    for (int t = 0; t < 12; ++t)
#pragma unroll
      for (int r = 0; r < 8; ++r) o[t][r] *= scale[r];

    // P (A-layout via LDS) x V (transposed -> contiguous B-frag rows)
    const v16bf pf = afrag(&parr[w][0][0], 32, 0, 0);
#pragma unroll
    for (int t = 0; t < 12; ++t)
      o[t] = wmma_bf16(pf, bfrag(vt_p, TT, t * 16, j0), o[t]);
  }

  float inv_l[8];
#pragma unroll
  for (int r = 0; r < 8; ++r) inv_l[r] = 1.0f / lrun[r];

  for (int t = 0; t < 12; ++t) {
    const int e = h * DV + t * 16 + cl;
    for (int r = 0; r < 8; ++r) {
      const int row = q0 + r + hi8;
      ao[((size_t)b * TT + row) * EE + e] = f2bf(o[t][r] * inv_l[r]);
    }
  }
}

// ---------------------------------------------------------------- launch

extern "C" void kernel_launch(void* const* d_in, const int* in_sizes, int n_in,
                              void* d_out, int out_size, void* d_ws,
                              size_t ws_size, hipStream_t stream) {
  (void)in_sizes; (void)n_in; (void)out_size; (void)ws_size;
  const float* x = (const float*)d_in[0];
  const float* Wq = (const float*)d_in[1];
  const float* Wk = (const float*)d_in[2];
  const float* Wv = (const float*)d_in[3];
  const float* Wrk = (const float*)d_in[4];
  const float* rwb = (const float*)d_in[5];
  const float* rrb = (const float*)d_in[6];
  const float* We = (const float*)d_in[7];
  const float* be = (const float*)d_in[8];

  char* p = (char*)d_ws;
  auto take = [&](size_t n) {
    char* r = p;
    p += (n + 255) & ~(size_t)255;
    return r;
  };
  __bf16* xb = (__bf16*)take((size_t)4 * TT * CC * 2);
  __bf16* wqb = (__bf16*)take((size_t)512 * CC * 2);
  __bf16* wkb = (__bf16*)take((size_t)512 * CC * 2);
  __bf16* wvb = (__bf16*)take((size_t)EE * CC * 2);
  __bf16* wrkb = (__bf16*)take((size_t)512 * 192 * 2);
  __bf16* web = (__bf16*)take((size_t)EE * EE * 2);
  __bf16* peb = (__bf16*)take((size_t)MP * 192 * 2);
  __bf16* qcb = (__bf16*)take((size_t)32 * TT * DK * 2);
  __bf16* qrb = (__bf16*)take((size_t)32 * TT * DK * 2);
  __bf16* kbf = (__bf16*)take((size_t)32 * TT * DK * 2);
  __bf16* vtb = (__bf16*)take((size_t)32 * DV * TT * 2);
  __bf16* rkb = (__bf16*)take((size_t)HH * MP * DK * 2);
  __bf16* aob = (__bf16*)take((size_t)4 * TT * EE * 2);
  float* maxp = (float*)take(256);

  auto cast = [&](const float* s, __bf16* d, size_t n) {
    cast_bf16_kernel<<<(int)((n + 255) / 256), 256, 0, stream>>>(s, d, (int)n);
  };
  cast(x, xb, (size_t)4 * TT * CC);
  cast(Wq, wqb, (size_t)512 * CC);
  cast(Wk, wkb, (size_t)512 * CC);
  cast(Wv, wvb, (size_t)EE * CC);
  cast(Wrk, wrkb, (size_t)512 * 192);
  cast(We, web, (size_t)EE * EE);

  gamma_max_kernel<<<1, 256, 0, stream>>>(maxp);
  pe_fill_kernel<<<(MP * 192 + 255) / 256, 256, 0, stream>>>(peb, maxp);

  // q (both biased copies): [6144,1536] x [512,1536]^T
  gemm_bf16_kernel<<<dim3(512 / 128, 6144 / 128), 256, 0, stream>>>(
      xb, wqb, CC, MODE_QCB, qcb, qrb, nullptr, rwb, rrb);
  // k
  gemm_bf16_kernel<<<dim3(512 / 128, 6144 / 128), 256, 0, stream>>>(
      xb, wkb, CC, MODE_K, kbf, nullptr, nullptr, nullptr, nullptr);
  // v (stored transposed)
  gemm_bf16_kernel<<<dim3(EE / 128, 6144 / 128), 256, 0, stream>>>(
      xb, wvb, CC, MODE_VT, vtb, nullptr, nullptr, nullptr, nullptr);
  // r_k: [3072,192] x [512,192]^T  (pe row 3071 is zero -> rk row 3071 zero)
  gemm_bf16_kernel<<<dim3(512 / 128, MP / 128), 256, 0, stream>>>(
      peb, wrkb, 192, MODE_RK, rkb, nullptr, nullptr, nullptr, nullptr);

  // fused attention with relative shift + online softmax
  attn_rpe_kernel<<<dim3(TT / 128, 32), 256, 0, stream>>>(qcb, qrb, kbf, vtb,
                                                          rkb, aob);

  // output projection: [6144,1536] x [1536,1536]^T + be -> fp32 d_out
  gemm_bf16_kernel<<<dim3(EE / 128, 6144 / 128), 256, 0, stream>>>(
      aob, web, EE, MODE_FINAL, nullptr, nullptr, (float*)d_out, be, nullptr);
}
